// v8DetectionLoss_10230612099532
// MI455X (gfx1250) — compile-verified, matrix operand order checked by reading, not run on hardware
//
#include <hip/hip_runtime.h>
#include <math.h>

// ---------------- problem constants ----------------
constexpr int B_   = 16;
constexpr int NC_  = 80;
constexpr int RMAX = 16;
constexpr int M_   = 32;
constexpr int NO_  = NC_ + 4 * RMAX;       // 144
constexpr int A0_  = 6400, A1_ = 1600, A2_ = 400;
constexpr int A_   = A0_ + A1_ + A2_;      // 8400
constexpr int TOPK = 10;
constexpr float EPS_  = 1e-9f;
constexpr float EPS7  = 1e-7f;

typedef __attribute__((ext_vector_type(16))) _Float16 v16h;
typedef __attribute__((ext_vector_type(8)))  float    v8f;
typedef __attribute__((ext_vector_type(4)))  int      v4i;

#if defined(__gfx1250__) && \
    __has_builtin(__builtin_amdgcn_global_load_async_to_lds_b128) && \
    __has_builtin(__builtin_amdgcn_s_wait_asynccnt)
#define USE_ASYNC_LDS 1
typedef __attribute__((address_space(1))) v4i* gv4i_p;  // global int32x4*
typedef __attribute__((address_space(3))) v4i* lv4i_p;  // LDS    int32x4*
#else
#define USE_ASYNC_LDS 0
#endif

// ---------------- device helpers ----------------
__device__ __forceinline__ void anchor_info(int a, int& lvl, int& sp, int& hw,
                                            float& ax, float& ay, float& st) {
  if (a < A0_)             { lvl = 0; sp = a;             hw = A0_; int W = 80; ax = (sp % W) + 0.5f; ay = (sp / W) + 0.5f; st = 8.f;  }
  else if (a < A0_ + A1_)  { lvl = 1; sp = a - A0_;       hw = A1_; int W = 40; ax = (sp % W) + 0.5f; ay = (sp / W) + 0.5f; st = 16.f; }
  else                     { lvl = 2; sp = a - A0_ - A1_; hw = A2_; int W = 20; ax = (sp % W) + 0.5f; ay = (sp / W) + 0.5f; st = 32.f; }
}

__device__ __forceinline__ const float* feat_of(int lvl, const float* f0, const float* f1, const float* f2) {
  return (lvl == 0) ? f0 : ((lvl == 1) ? f1 : f2);
}

__device__ __forceinline__ float block_sum256(float v, float* sh) {
  int t = threadIdx.x;
  sh[t] = v; __syncthreads();
  for (int s = 128; s > 0; s >>= 1) { if (t < s) sh[t] += sh[t + s]; __syncthreads(); }
  float r = sh[0]; __syncthreads();
  return r;
}

// ---------------- K1: DFL decode via WMMA ----------------
// Each wave handles 16 (b,a,side) rows: softmax over 16 bins, then one
// v_wmma_f32_16x16x32_f16 computing probs @ proj (proj broadcast into all
// 16 B-columns; K=16..31 zero padded). 8 waves/block, 128 rows/block.
__global__ __launch_bounds__(256)
void k_decode(const float* __restrict__ f0, const float* __restrict__ f1,
              const float* __restrict__ f2, float* __restrict__ pbox) {
  const int lane  = threadIdx.x & 31;
  const int wv    = threadIdx.x >> 5;
  const int gw    = blockIdx.x * 8 + wv;        // global wave id
  const int rg    = gw * 16;                    // first row of this wave's tile
  const int b     = rg / (A_ * 4);
  const int aBase = (rg % (A_ * 4)) / 4;        // 16 rows == 4 anchors (aligned)

  const int m     = lane & 15;                  // WMMA row index this lane feeds
  const int kbase = (lane < 16) ? 0 : 8;        // A-matrix 16-bit layout halves
  const int a     = aBase + (m >> 2);
  const int side  = m & 3;

  int lvl, sp, hw; float ax, ay, st;
  anchor_info(a, lvl, sp, hw, ax, ay, st);
  const float* f = feat_of(lvl, f0, f1, f2);
  const float* base = f + (long)(b * NO_ + side * RMAX + kbase) * hw + sp;

  float x[8];
#pragma unroll
  for (int j = 0; j < 8; ++j) x[j] = base[(long)j * hw];

  // row softmax: lanes L and L+16 hold the two halves of the same row
  float mx = x[0];
#pragma unroll
  for (int j = 1; j < 8; ++j) mx = fmaxf(mx, x[j]);
  mx = fmaxf(mx, __shfl_xor(mx, 16, 32));
  float e[8], s = 0.f;
#pragma unroll
  for (int j = 0; j < 8; ++j) { e[j] = __expf(x[j] - mx); s += e[j]; }
  s += __shfl_xor(s, 16, 32);
  const float inv = 1.f / s;

  union { v16h v; _Float16 h[16]; } ua = {}, ub = {};
#pragma unroll
  for (int j = 0; j < 8; ++j) {
    ua.h[j] = (_Float16)(e[j] * inv);           // A: probs, K(pad)=0
    ub.h[j] = (_Float16)(float)(kbase + j);     // B: proj broadcast to all cols
  }

  v8f c = {};
  union { v8f v; float f_[8]; } ud;
  ud.v = __builtin_amdgcn_wmma_f32_16x16x32_f16(
      /*neg_a=*/false, ua.v, /*neg_b=*/false, ub.v,
      /*c_mod=*/(short)0, c, /*reuse_a=*/false, /*reuse_b=*/false);

  // D layout: VGPR r -> (lanes 0-15: M=r, N=lane) (lanes 16-31: M=8+r).
  // Lane 0 holds rows 0..7 (col 0), lane 16 holds rows 8..15 (col 0).
  if (lane == 0 || lane == 16) {
    const int afirst = aBase + ((lane == 0) ? 0 : 2);
#pragma unroll
    for (int q = 0; q < 2; ++q) {
      const int aa = afirst + q;
      int l2, s2, h2; float ax2, ay2, st2;
      anchor_info(aa, l2, s2, h2, ax2, ay2, st2);
      const float d0 = ud.f_[4 * q + 0], d1 = ud.f_[4 * q + 1];
      const float d2 = ud.f_[4 * q + 2], d3 = ud.f_[4 * q + 3];
      reinterpret_cast<float4*>(pbox)[(long)b * A_ + aa] =
          make_float4(ax2 - d0, ay2 - d1, ax2 + d2, ay2 + d3);
    }
  }
}

// ---------------- K2: pairwise overlaps + align metric ----------------
__global__ __launch_bounds__(256)
void k_pair(const float* __restrict__ f0, const float* __restrict__ f1,
            const float* __restrict__ f2, const float* __restrict__ pbox,
            const float* __restrict__ gtb, const int* __restrict__ gtl,
            const float* __restrict__ mgt,
            float* __restrict__ align, float* __restrict__ ovl) {
  const long idx = (long)blockIdx.x * 256 + threadIdx.x;
  if (idx >= (long)B_ * M_ * A_) return;
  const int b = (int)(idx / (M_ * A_));
  const int r = (int)(idx % (M_ * A_));
  const int m = r / A_, a = r % A_;
  const int gi = b * M_ + m;

  const float mg  = mgt[gi];
  const float gx1 = gtb[gi * 4 + 0], gy1 = gtb[gi * 4 + 1];
  const float gx2 = gtb[gi * 4 + 2], gy2 = gtb[gi * 4 + 3];
  const int   cls = gtl[gi];

  int lvl, sp, hw; float ax, ay, st;
  anchor_info(a, lvl, sp, hw, ax, ay, st);
  const float apx = ax * st, apy = ay * st;

  const bool inb = (apx - gx1 > EPS_) && (apy - gy1 > EPS_) &&
                   (gx2 - apx > EPS_) && (gy2 - apy > EPS_);
  const bool valid = inb && (mg > 0.f);

  float al = 0.f, ov = 0.f;
  if (valid) {
    const float* f = feat_of(lvl, f0, f1, f2);
    const float* sc_p = &f[(long)(b * NO_ + 4 * RMAX + cls) * hw + sp];
    __builtin_prefetch(sc_p, 0, 0);   // global_prefetch_b8
    const float xs = *sc_p;
    const float sc = 1.f / (1.f + __expf(-xs));

    const float4 pb = reinterpret_cast<const float4*>(pbox)[(long)b * A_ + a];
    const float px1 = pb.x * st, py1 = pb.y * st, px2 = pb.z * st, py2 = pb.w * st;
    const float iw = fmaxf(fminf(px2, gx2) - fmaxf(px1, gx1), 0.f);
    const float ih = fmaxf(fminf(py2, gy2) - fmaxf(py1, gy1), 0.f);
    const float inter = iw * ih;
    const float uni = (px2 - px1) * (py2 - py1) + (gx2 - gx1) * (gy2 - gy1) - inter + EPS7;
    ov = inter / uni;
    const float o2 = ov * ov;
    al = sqrtf(sc) * (o2 * o2 * o2);            // sc^0.5 * ov^6
  }
  align[(long)gi * A_ + a] = al;
  ovl[(long)gi * A_ + a]   = ov;
}

// ---------------- K3: top-10 per (b,m) over anchors ----------------
// Stage the 33.6KB align row into LDS once (CDNA5 async global->LDS copy,
// ASYNCcnt-tracked, when available), then run all 10 extraction passes
// from LDS instead of re-reading L2 ten times.
__global__ __launch_bounds__(256)
void k_topk(const float* __restrict__ align, const float* __restrict__ mgt,
            unsigned char* __restrict__ mtopk) {
  __shared__ __align__(16) float srow[A_];     // 33,600 B of the 320KB WGP pool
  __shared__ float shv[256];
  __shared__ int   shi[256];
  __shared__ int   sel[TOPK];
  const int gi  = blockIdx.x;          // (b,m)
  const int tid = threadIdx.x;
  const float* row = align + (long)gi * A_;
  unsigned char* mrow = mtopk + (long)gi * A_;

  for (int a = tid; a < A_; a += 256) mrow[a] = 0;
  __syncthreads();
  if (mgt[gi] <= 0.f) return;          // uniform exit per block

#if USE_ASYNC_LDS
  // per-lane async b128 descriptors: global row -> LDS, no VGPR round-trip
  for (int c4 = tid; c4 < A_ / 4; c4 += 256) {
    __builtin_amdgcn_global_load_async_to_lds_b128(
        (gv4i_p)(row + 4 * c4),
        (lv4i_p)(&srow[4 * c4]),
        /*offset=*/0, /*cpol=*/0);
  }
  __builtin_amdgcn_s_wait_asynccnt(0);         // s_wait_asynccnt 0
#else
  for (int a = tid; a < A_; a += 256) srow[a] = row[a];
#endif
  __syncthreads();

  for (int p = 0; p < TOPK; ++p) {
    float bv = -1.f; int bi = 0x7fffffff;
    for (int a = tid; a < A_; a += 256) {
      bool ex = false;
      for (int t = 0; t < p; ++t) if (sel[t] == a) { ex = true; break; }
      if (ex) continue;
      const float v = srow[a];
      if (v > bv) { bv = v; bi = a; }
    }
    shv[tid] = bv; shi[tid] = bi; __syncthreads();
    for (int s = 128; s > 0; s >>= 1) {
      if (tid < s) {
        if (shv[tid + s] > shv[tid] ||
            (shv[tid + s] == shv[tid] && shi[tid + s] < shi[tid])) {
          shv[tid] = shv[tid + s]; shi[tid] = shi[tid + s];
        }
      }
      __syncthreads();
    }
    if (tid == 0) { sel[p] = shi[0]; if (shi[0] < A_) mrow[shi[0]] = 1; }
    __syncthreads();
  }
}

// ---------------- K4: resolve mask_pos per (b,a) ----------------
__global__ __launch_bounds__(256)
void k_select(const float* __restrict__ gtb, const int* __restrict__ gtl,
              const float* __restrict__ mgt,
              const unsigned char* __restrict__ mtopk,
              const float* __restrict__ ovl,
              unsigned int* __restrict__ bits, float* __restrict__ fgm,
              int* __restrict__ tgt) {
  const long idx = (long)blockIdx.x * 256 + threadIdx.x;
  if (idx >= (long)B_ * A_) return;
  const int b = (int)(idx / A_), a = (int)(idx % A_);

  int lvl, sp, hw; float ax, ay, st;
  anchor_info(a, lvl, sp, hw, ax, ay, st);
  const float apx = ax * st, apy = ay * st;

  unsigned int bm = 0; int fg = 0;
  for (int m = 0; m < M_; ++m) {
    const int gi = b * M_ + m;
    if (!mtopk[(long)gi * A_ + a]) continue;
    const float gx1 = gtb[gi * 4 + 0], gy1 = gtb[gi * 4 + 1];
    const float gx2 = gtb[gi * 4 + 2], gy2 = gtb[gi * 4 + 3];
    const bool inb = (apx - gx1 > EPS_) && (apy - gy1 > EPS_) &&
                     (gx2 - apx > EPS_) && (gy2 - apy > EPS_);
    if (inb && mgt[gi] > 0.f) { bm |= 1u << m; ++fg; }
  }

  float fgv; int t;
  if (fg > 1) {          // conflict: keep GT with max overlap (first-max tie)
    float bo = -1.f; int bi2 = 0;
    for (int m = 0; m < M_; ++m) {
      const float o = ovl[(long)(b * M_ + m) * A_ + a];
      if (o > bo) { bo = o; bi2 = m; }
    }
    bm = 1u << bi2; fgv = 1.f; t = bi2;
  } else {
    fgv = (float)fg;
    t = bm ? (__ffs(bm) - 1) : 0;
  }
  bits[idx] = bm; fgm[idx] = fgv; tgt[idx] = t;
}

// ---------------- K5: per-(b,m) max of masked align / overlap ----------------
__global__ __launch_bounds__(256)
void k_posmax(const float* __restrict__ align, const float* __restrict__ ovl,
              const unsigned int* __restrict__ bits,
              float* __restrict__ pal, float* __restrict__ pov) {
  __shared__ float sa[256], so[256];
  const int gi = blockIdx.x;            // (b,m)
  const int b = gi / M_, m = gi % M_;
  const int tid = threadIdx.x;
  float ma = 0.f, mo = 0.f;
  for (int a = tid; a < A_; a += 256) {
    if ((bits[(long)b * A_ + a] >> m) & 1u) {
      ma = fmaxf(ma, align[(long)gi * A_ + a]);
      mo = fmaxf(mo, ovl[(long)gi * A_ + a]);
    }
  }
  sa[tid] = ma; so[tid] = mo; __syncthreads();
  for (int s = 128; s > 0; s >>= 1) {
    if (tid < s) { sa[tid] = fmaxf(sa[tid], sa[tid + s]); so[tid] = fmaxf(so[tid], so[tid + s]); }
    __syncthreads();
  }
  if (tid == 0) { pal[gi] = sa[0]; pov[gi] = so[0]; }
}

// ---------------- K6: per-anchor fg losses (tss, bce-fg, ciou, dfl) ----------
__global__ __launch_bounds__(256)
void k_loss_fg(const float* __restrict__ f0, const float* __restrict__ f1,
               const float* __restrict__ f2, const float* __restrict__ pbox,
               const float* __restrict__ gtb, const int* __restrict__ gtl,
               const float* __restrict__ align,
               const unsigned int* __restrict__ bits,
               const float* __restrict__ fgm, const int* __restrict__ tgt,
               const float* __restrict__ pal, const float* __restrict__ pov,
               float* __restrict__ sums) {
  __shared__ float sh[256];
  const long idx = (long)blockIdx.x * 256 + threadIdx.x;
  const bool in = idx < (long)B_ * A_;

  float tssC = 0.f, bceC = 0.f, iouC = 0.f, dflC = 0.f;
  if (in) {
    const int b = (int)(idx / A_), a = (int)(idx % A_);
    const unsigned int bm = bits[idx];
    const float fg = fgm[idx];
    const int   t  = tgt[idx];

    // norm = max_m am * pos_ovl / (pos_align + eps)
    float norm = 0.f;
    unsigned int bb = bm;
    while (bb) {
      const int m = __ffs(bb) - 1; bb &= bb - 1;
      const int gi = b * M_ + m;
      const float am = align[(long)gi * A_ + a];
      norm = fmaxf(norm, am * pov[gi] / (pal[gi] + EPS_));
    }

    if (fg > 0.f) {
      int lvl, sp, hw; float ax, ay, st;
      anchor_info(a, lvl, sp, hw, ax, ay, st);
      const float* f = feat_of(lvl, f0, f1, f2);
      const int gi = b * M_ + t;
      int lab = gtl[gi]; lab = lab < 0 ? 0 : lab;
      const float inv_st = 1.f / st;
      const float tx1 = gtb[gi * 4 + 0] * inv_st, ty1 = gtb[gi * 4 + 1] * inv_st;
      const float tx2 = gtb[gi * 4 + 2] * inv_st, ty2 = gtb[gi * 4 + 3] * inv_st;
      const float weight = fmaxf(norm, 1e-6f);

      tssC = norm;
      bceC = -f[(long)(b * NO_ + 4 * RMAX + lab) * hw + sp] * norm;

      // ----- CIoU (unscaled pred vs target/stride) -----
      const float4 pb = reinterpret_cast<const float4*>(pbox)[(long)b * A_ + a];
      const float w1 = pb.z - pb.x, h1 = pb.w - pb.y + EPS7;
      const float w2 = tx2 - tx1,   h2 = ty2 - ty1 + EPS7;
      const float iw = fmaxf(fminf(pb.z, tx2) - fmaxf(pb.x, tx1), 0.f);
      const float ih = fmaxf(fminf(pb.w, ty2) - fmaxf(pb.y, ty1), 0.f);
      const float inter = iw * ih;
      const float uni = w1 * h1 + w2 * h2 - inter + EPS7;
      const float iou = inter / uni;
      const float cw = fmaxf(pb.z, tx2) - fminf(pb.x, tx1);
      const float ch = fmaxf(pb.w, ty2) - fminf(pb.y, ty1);
      const float c2 = cw * cw + ch * ch + EPS7;
      const float dx = tx1 + tx2 - pb.x - pb.z, dy = ty1 + ty2 - pb.y - pb.w;
      const float rho2 = (dx * dx + dy * dy) * 0.25f;
      const float da = atanf(w2 / (h2 + EPS7)) - atanf(w1 / (h1 + EPS7));
      const float v = (4.f / (float)(M_PI * M_PI)) * da * da;
      const float alpha = v / (v - iou + 1.f + EPS7);
      float ci = iou - (rho2 / c2 + v * alpha);
      ci = fminf(fmaxf(ci, -1.f), 1.f);
      iouC = (1.f - ci) * weight * fg;

      // ----- DFL -----
      const float tgt_ltrb[4] = { ax - tx1, ay - ty1, tx2 - ax, ty2 - ay };
      float acc = 0.f;
#pragma unroll
      for (int side = 0; side < 4; ++side) {
        float tv = fminf(fmaxf(tgt_ltrb[side], 0.f), (float)RMAX - 1.01f);
        int tl = (int)tv; tl = tl > RMAX - 2 ? RMAX - 2 : tl;
        const float wr = fminf(fmaxf(tv - (float)tl, 0.f), 1.f);
        const float wl = fminf(fmaxf((float)(tl + 1) - tv, 0.f), 1.f);
        const float* base = f + (long)(b * NO_ + side * RMAX) * hw + sp;
        float xv[RMAX], mxv = -1e30f;
#pragma unroll
        for (int k = 0; k < RMAX; ++k) { xv[k] = base[(long)k * hw]; mxv = fmaxf(mxv, xv[k]); }
        float se = 0.f;
#pragma unroll
        for (int k = 0; k < RMAX; ++k) se += __expf(xv[k] - mxv);
        const float lse = mxv + __logf(se);
        acc += (lse - xv[tl]) * wl + (lse - xv[tl + 1]) * wr;
      }
      dflC = fminf(acc * 0.25f, 100.f) * weight * fg;
    }
  }

  float r;
  r = block_sum256(bceC, sh); if (threadIdx.x == 0) atomicAdd(&sums[0], r);
  r = block_sum256(tssC, sh); if (threadIdx.x == 0) atomicAdd(&sums[1], r);
  r = block_sum256(iouC, sh); if (threadIdx.x == 0) atomicAdd(&sums[2], r);
  r = block_sum256(dflC, sh); if (threadIdx.x == 0) atomicAdd(&sums[3], r);
}

// ---------------- K7: dense BCE (softplus part) over all class logits -------
__global__ __launch_bounds__(256)
void k_bce_dense(const float* __restrict__ f0, const float* __restrict__ f1,
                 const float* __restrict__ f2, float* __restrict__ sums) {
  __shared__ float sh[256];
  constexpr long N0 = (long)B_ * NC_ * A0_;   // 8,192,000
  constexpr long N1 = (long)B_ * NC_ * A1_;   // 2,048,000
  constexpr long N2 = (long)B_ * NC_ * A2_;   //   512,000
  constexpr long NT = N0 + N1 + N2;

  const long base = (long)blockIdx.x * 1024 + threadIdx.x;
  float acc = 0.f;
#pragma unroll
  for (int i = 0; i < 4; ++i) {
    const long id = base + (long)i * 256;
    if (id >= NT) break;
    const float* f; long id2; int hw;
    if (id < N0)           { f = f0; id2 = id;            hw = A0_; }
    else if (id < N0 + N1) { f = f1; id2 = id - N0;       hw = A1_; }
    else                   { f = f2; id2 = id - N0 - N1;  hw = A2_; }
    const int sp = (int)(id2 % hw);
    const long tt = id2 / hw;
    const int cc = (int)(tt % NC_);
    const int b  = (int)(tt / NC_);
    const float* p = &f[(long)(b * NO_ + 4 * RMAX + cc) * hw + sp];
    if (i == 0) __builtin_prefetch(p + 1024, 0, 0);     // global_prefetch_b8
    const float x = *p;
    acc += fmaxf(x, 0.f) + log1pf(__expf(-fabsf(x)));
  }
  const float r = block_sum256(acc, sh);
  if (threadIdx.x == 0) atomicAdd(&sums[0], r);
}

// ---------------- K0 / K8 ----------------
__global__ void k_zero(float* sums) { if (threadIdx.x < 8) sums[threadIdx.x] = 0.f; }

__global__ void k_final(const float* __restrict__ sums, float* __restrict__ out) {
  if (threadIdx.x == 0 && blockIdx.x == 0) {
    __builtin_amdgcn_s_wait_tensorcnt(0);   // CDNA5 split-counter wait (no-op here)
    const float tss  = fmaxf(sums[1], 1.f);
    const float lcls = sums[0] / tss;
    const float liou = fminf(sums[2] / tss, 100.f);
    const float ldfl = fminf(sums[3] / tss, 100.f);
    out[0] = (7.5f * liou + 0.5f * lcls + 1.5f * ldfl) * (float)B_;
  }
}

// ---------------- host launcher ----------------
extern "C" void kernel_launch(void* const* d_in, const int* in_sizes, int n_in,
                              void* d_out, int out_size, void* d_ws, size_t ws_size,
                              hipStream_t stream) {
  const float* f0  = (const float*)d_in[0];
  const float* f1  = (const float*)d_in[1];
  const float* f2  = (const float*)d_in[2];
  const float* gtb = (const float*)d_in[3];
  const int*   gtl = (const int*)  d_in[4];
  const float* mgt = (const float*)d_in[5];
  float* out = (float*)d_out;

  // workspace carve-up (256B aligned); total ~43 MB
  char* w = (char*)d_ws;
  size_t off = 0;
  auto carve = [&](size_t bytes) { void* p = w + off; off = (off + bytes + 255) & ~(size_t)255; return p; };
  float*         sums  = (float*)        carve(8 * sizeof(float));
  float*         pbox  = (float*)        carve((size_t)B_ * A_ * 4 * sizeof(float));
  float*         align = (float*)        carve((size_t)B_ * M_ * A_ * sizeof(float));
  float*         ovl   = (float*)        carve((size_t)B_ * M_ * A_ * sizeof(float));
  unsigned char* mtopk = (unsigned char*)carve((size_t)B_ * M_ * A_);
  unsigned int*  bits  = (unsigned int*) carve((size_t)B_ * A_ * sizeof(unsigned int));
  float*         fgm   = (float*)        carve((size_t)B_ * A_ * sizeof(float));
  int*           tgt   = (int*)          carve((size_t)B_ * A_ * sizeof(int));
  float*         pal   = (float*)        carve((size_t)B_ * M_ * sizeof(float));
  float*         pov   = (float*)        carve((size_t)B_ * M_ * sizeof(float));
  (void)ws_size; (void)in_sizes; (void)n_in; (void)out_size;

  k_zero<<<1, 32, 0, stream>>>(sums);

  // K1: B*A*4 rows / (8 waves * 16 rows) = 4200 blocks
  k_decode<<<(B_ * A_ * 4) / 128, 256, 0, stream>>>(f0, f1, f2, pbox);

  // K2: B*M*A element grid
  k_pair<<<(B_ * M_ * A_ + 255) / 256, 256, 0, stream>>>(f0, f1, f2, pbox, gtb, gtl, mgt, align, ovl);

  // K3: one block per (b,m), align row staged to LDS (async path on CDNA5)
  k_topk<<<B_ * M_, 256, 0, stream>>>(align, mgt, mtopk);

  // K4: per-anchor resolution
  k_select<<<(B_ * A_ + 255) / 256, 256, 0, stream>>>(gtb, gtl, mgt, mtopk, ovl, bits, fgm, tgt);

  // K5: one block per (b,m)
  k_posmax<<<B_ * M_, 256, 0, stream>>>(align, ovl, bits, pal, pov);

  // K6: per-anchor fg losses
  k_loss_fg<<<(B_ * A_ + 255) / 256, 256, 0, stream>>>(f0, f1, f2, pbox, gtb, gtl,
                                                       align, bits, fgm, tgt, pal, pov, sums);

  // K7: dense BCE softplus over B*NC*A logits, 4 elems/thread
  const long nbce = (long)B_ * NC_ * A_;
  k_bce_dense<<<(unsigned)((nbce + 1023) / 1024), 256, 0, stream>>>(f0, f1, f2, sums);

  k_final<<<1, 32, 0, stream>>>(sums, out);
}
